// SignalTransformerEncoder_80977313399483
// MI455X (gfx1250) — matrix operand order repaired and checked by reading, not compiled, not run on hardware
//
#include <hip/hip_runtime.h>
#include <hip/hip_bf16.h>

typedef __attribute__((ext_vector_type(16))) __bf16 v16bf;
typedef __attribute__((ext_vector_type(8)))  float  v8f;

// ---------------- helpers ----------------
__device__ __forceinline__ unsigned short f2bf(float f) {
    unsigned int u = __float_as_uint(f);
    u += 0x7FFFu + ((u >> 16) & 1u);           // round-to-nearest-even
    return (unsigned short)(u >> 16);
}
__device__ __forceinline__ float gelu_exact(float v) {
    return 0.5f * v * (1.0f + erff(v * 0.70710678118654752f));
}

// ---------------- conv1d (direct) ----------------
__global__ __launch_bounds__(256) void conv1d_k(
    const float* __restrict__ x, const float* __restrict__ w,
    const float* __restrict__ bias, float* __restrict__ y,
    int B, int Cin, int Lin, int Cout, int Lout, int Ksz, int stride, int pad)
{
    long idx = (long)blockIdx.x * 256 + threadIdx.x;
    long total = (long)B * Cout * Lout;
    if (idx >= total) return;
    int lo = (int)(idx % Lout);
    long t = idx / Lout;
    int co = (int)(t % Cout);
    int b  = (int)(t / Cout);
    float acc = bias[co];
    int start = lo * stride - pad;
    const float* wp = w + (size_t)co * Cin * Ksz;
    for (int ci = 0; ci < Cin; ++ci) {
        const float* xp = x + ((size_t)b * Cin + ci) * Lin;
        const float* wc = wp + (size_t)ci * Ksz;
        for (int kk = 0; kk < Ksz; ++kk) {
            int p = start + kk;
            if (p >= 0 && p < Lin) acc += xp[p] * wc[kk];
        }
    }
    y[idx] = acc;
}

// ---------------- groupnorm (1 group, per batch) ----------------
__global__ __launch_bounds__(256) void gn_reduce_k(
    const float* __restrict__ x, float* __restrict__ stats, long n, float eps)
{
    __shared__ float ssum[256], ssq[256];
    const float* xb = x + (size_t)blockIdx.x * n;
    float s = 0.f, q = 0.f;
    for (long i = threadIdx.x; i < n; i += 256) { float v = xb[i]; s += v; q += v * v; }
    ssum[threadIdx.x] = s; ssq[threadIdx.x] = q;
    __syncthreads();
    for (int o = 128; o > 0; o >>= 1) {
        if ((int)threadIdx.x < o) { ssum[threadIdx.x] += ssum[threadIdx.x + o]; ssq[threadIdx.x] += ssq[threadIdx.x + o]; }
        __syncthreads();
    }
    if (threadIdx.x == 0) {
        float mu  = ssum[0] / (float)n;
        float var = ssq[0] / (float)n - mu * mu;
        stats[blockIdx.x * 2 + 0] = mu;
        stats[blockIdx.x * 2 + 1] = rsqrtf(var + eps);
    }
}

__global__ __launch_bounds__(256) void gn_apply_gelu_k(
    float* __restrict__ x, const float* __restrict__ stats,
    const float* __restrict__ g, const float* __restrict__ be, int C, int L, long total)
{
    long idx = (long)blockIdx.x * 256 + threadIdx.x;
    if (idx >= total) return;
    int c = (int)((idx / L) % C);
    int b = (int)(idx / ((long)C * L));
    float mu = stats[b * 2 + 0], rs = stats[b * 2 + 1];
    float v = (x[idx] - mu) * rs * g[c] + be[c];
    x[idx] = gelu_exact(v);
}

// ---------------- transpose (B,C,L)->(B,L,C) + positional enc ----------------
__global__ __launch_bounds__(256) void transpose_pe_k(
    const float* __restrict__ y, float* __restrict__ h, int B, int D, int S)
{
    long idx = (long)blockIdx.x * 256 + threadIdx.x;
    long total = (long)B * S * D;
    if (idx >= total) return;
    int d = (int)(idx % D);
    int s = (int)((idx / D) % S);
    int b = (int)(idx / ((long)S * D));
    int i = d >> 1;
    float div = __expf(-(float)(2 * i) * 9.210340371976184f / (float)D); // ln(10000)
    float ang = (float)b * div;                 // pe[:B] broadcasts row b over all s
    float pe = (d & 1) ? __cosf(ang) : __sinf(ang);
    h[idx] = y[((size_t)b * D + d) * S + s] + pe;
}

// ---------------- f32 -> bf16 ----------------
__global__ __launch_bounds__(256) void tobf16_k(
    const float* __restrict__ x, unsigned short* __restrict__ y, long n)
{
    long idx = (long)blockIdx.x * 256 + threadIdx.x;
    if (idx < n) y[idx] = f2bf(x[idx]);
}

// ---------------- bf16 WMMA GEMM ----------------
// out[m,n] = act((sum_k A[m,k]*W[n,k] + bias[n]) * scale)
// A: M x K bf16 (K contiguous), W: N x K bf16 (K contiguous).
// Block: 256 threads = 8 waves stacked along M; wave tile = 32(M) x 32(N) = 2x2 WMMA tiles.
// Block tile: 256(M) x 32(N). The 32-column W K-panel is shared by all 8 waves and is
// staged through LDS in 800-K chunks via GLOBAL_LOAD_ASYNC_TO_LDS_B128 (ASYNCcnt).
#define GEMM_KC 800
__global__ __launch_bounds__(256) void gemm_bf16_wmma_k(
    const unsigned short* __restrict__ A, const unsigned short* __restrict__ W,
    const float* __restrict__ bias, float* __restrict__ outF,
    unsigned short* __restrict__ outB, int M, int N, int K, float scale, int act)
{
    __shared__ unsigned short sW[32 * GEMM_KC];          // 50 KB W panel (32 cols x 800 K)

    const int tid  = threadIdx.x;
    const int wave = tid >> 5;             // 0..7, stacked along M
    const int lane = tid & 31;
    const int mBase = blockIdx.x * 256 + wave * 32;
    const int nBase = blockIdx.y * 32;
    const int hi = lane >> 4;              // lane half
    const int lm = lane & 15;

    // CDNA5 WMMA 16x16x32 bf16 fragment layouts (ISA 7.12.2):
    //  A (16x32): lane half hi, row m=lm; vals[0..7]=K 8*hi+j, vals[8..15]=K 16+8*hi+j
    //  B (32x16): lane half hi, col n=lm; vals[0..7]=K 8*hi+j, vals[8..15]=K 16+8*hi+j
    const unsigned short* aRow0 = A + (size_t)(mBase + lm) * K;
    const unsigned short* aRow1 = A + (size_t)(mBase + 16 + lm) * K;

    v8f acc[2][2] = {};
    for (int kc = 0; kc < K; kc += GEMM_KC) {
        __syncthreads();                                  // protect sW from prior readers
        // cooperative async copy: 32 rows x 1600 B = 100 x 16B chunks per row
        for (int i = tid; i < 32 * (GEMM_KC / 8); i += 256) {
            int nl = i / (GEMM_KC / 8);
            int c  = i % (GEMM_KC / 8);
            unsigned long long ga =
                (unsigned long long)(const void*)(W + (size_t)(nBase + nl) * K + kc + c * 8);
            unsigned la = (unsigned)(unsigned long long)(const void*)&sW[nl * GEMM_KC + c * 8];
            asm volatile("global_load_async_to_lds_b128 %0, %1, off"
                         :: "v"(la), "v"(ga) : "memory");
        }
        asm volatile("s_wait_asynccnt 0x0" ::: "memory"); // drain this wave's async copies
        __syncthreads();                                  // panel visible to all waves

        for (int k0 = 0; k0 < GEMM_KC; k0 += 32) {
            union { v16bf v; uint4 q[2]; } af[2], bf[2];
            af[0].q[0] = *(const uint4*)(aRow0 + kc + k0 + 8 * hi);
            af[0].q[1] = *(const uint4*)(aRow0 + kc + k0 + 16 + 8 * hi);
            af[1].q[0] = *(const uint4*)(aRow1 + kc + k0 + 8 * hi);
            af[1].q[1] = *(const uint4*)(aRow1 + kc + k0 + 16 + 8 * hi);
            __builtin_prefetch((const void*)(aRow0 + kc + k0 + 256), 0, 1); // global_prefetch_b8
            __builtin_prefetch((const void*)(aRow1 + kc + k0 + 256), 0, 1);
            #pragma unroll
            for (int t = 0; t < 2; ++t) {
                const unsigned short* wp = &sW[(t * 16 + lm) * GEMM_KC + k0 + 8 * hi];
                bf[t].q[0] = *(const uint4*)(wp);         // ds_load_b128
                bf[t].q[1] = *(const uint4*)(wp + 16);
            }
            #pragma unroll
            for (int r = 0; r < 2; ++r)
                #pragma unroll
                for (int t = 0; t < 2; ++t)
                    acc[r][t] = __builtin_amdgcn_wmma_f32_16x16x32_bf16(
                        false, af[r].v, false, bf[t].v, (short)0, acc[r][t], false, false);
        }
    }

    #pragma unroll
    for (int t = 0; t < 2; ++t) {
        const int n = nBase + t * 16 + lm;
        const float bv = bias ? bias[n] : 0.0f;
        #pragma unroll
        for (int r = 0; r < 2; ++r) {
            #pragma unroll
            for (int rr = 0; rr < 8; ++rr) {
                int m = mBase + r * 16 + rr + 8 * hi;     // C/D: VGPR rr -> row rr+8*hi
                float v = (acc[r][t][rr] + bv) * scale;
                if (act == 1) v = gelu_exact(v);
                size_t o = (size_t)m * N + n;
                if (outF) outF[o] = v;
                if (outB) outB[o] = f2bf(v);
            }
        }
    }
}

// ---------------- residual add + layernorm over last dim ----------------
__global__ __launch_bounds__(256) void add_ln_k(
    float* __restrict__ h, const float* __restrict__ r,
    const float* __restrict__ g, const float* __restrict__ be, int D)
{
    __shared__ float ssum[256], ssq[256];
    __shared__ float smu, srs;
    float* hp = h + (size_t)blockIdx.x * D;
    const float* rp = r + (size_t)blockIdx.x * D;
    float s = 0.f, q = 0.f;
    for (int i = threadIdx.x; i < D; i += 256) { float v = hp[i] + rp[i]; s += v; q += v * v; }
    ssum[threadIdx.x] = s; ssq[threadIdx.x] = q;
    __syncthreads();
    for (int o = 128; o > 0; o >>= 1) {
        if ((int)threadIdx.x < o) { ssum[threadIdx.x] += ssum[threadIdx.x + o]; ssq[threadIdx.x] += ssq[threadIdx.x + o]; }
        __syncthreads();
    }
    if (threadIdx.x == 0) {
        float mu = ssum[0] / (float)D;
        float var = ssq[0] / (float)D - mu * mu;
        smu = mu; srs = rsqrtf(var + 1e-12f);
    }
    __syncthreads();
    float mu = smu, rs = srs;
    for (int i = threadIdx.x; i < D; i += 256) {
        float v = hp[i] + rp[i];
        hp[i] = (v - mu) * rs * g[i] + be[i];
    }
}

// ---------------- sliding-window attention (w=8, 3w=24 keys/query) ----------------
#define WATT 8
__global__ __launch_bounds__(256) void attn_k(
    const float* __restrict__ q, const float* __restrict__ k, const float* __restrict__ v,
    float* __restrict__ o, int B, int S, int H, int D)
{
    __shared__ float sc[8][24];
    const int wave = threadIdx.x >> 5, lane = threadIdx.x & 31;
    long row = (long)blockIdx.x * 8 + wave;       // (b,h,c,qi)
    const int nc = S / WATT;
    int qi = (int)(row % WATT); long t = row / WATT;
    int c = (int)(t % nc); t /= nc;
    int hh = (int)(t % H);  int b = (int)(t / H);
    int sQ = c * WATT + qi;
    const float* qp = q + (((size_t)b * S + sQ) * H + hh) * D;
    if (lane < 24) {
        int kpos = c * WATT + lane - WATT;
        int dlt = lane - WATT - qi;
        float sv = -1e9f;                         // NEG_BIG
        if (dlt >= -WATT && dlt <= WATT && kpos >= 0 && kpos < S) {
            const float* kp = k + (((size_t)b * S + kpos) * H + hh) * D;
            float a = 0.f;
            for (int d = 0; d < D; ++d) a += qp[d] * kp[d];   // q pre-scaled by 1/sqrt(D)
            sv = a;
        }
        sc[wave][lane] = sv;
    }
    __syncthreads();
    float mx = -1e30f;
    #pragma unroll
    for (int j = 0; j < 24; ++j) mx = fmaxf(mx, sc[wave][j]);
    float sum = 0.f;
    #pragma unroll
    for (int j = 0; j < 24; ++j) sum += __expf(sc[wave][j] - mx);
    float inv = 1.0f / sum;
    float* op = o + (((size_t)b * S + sQ) * H + hh) * D;
    for (int d = lane; d < D; d += 32) {
        float a = 0.f;
        for (int j = 0; j < 24; ++j) {
            int kpos = c * WATT + j - WATT;
            if (kpos >= 0 && kpos < S) {
                float p = __expf(sc[wave][j] - mx) * inv;
                a += p * v[(((size_t)b * S + kpos) * H + hh) * D + d];
            }
        }
        op[d] = a;
    }
}

// ---------------- host orchestration ----------------
extern "C" void kernel_launch(void* const* d_in, const int* in_sizes, int n_in,
                              void* d_out, int out_size, void* d_ws, size_t ws_size,
                              hipStream_t stream)
{
    // input order per setup_inputs(): src, then params leaves in dict order
    const float* src     = (const float*)d_in[0];
    const float* conv1_w = (const float*)d_in[1];
    const float* conv1_b = (const float*)d_in[2];
    const float* gn1_g   = (const float*)d_in[3];
    const float* gn1_b   = (const float*)d_in[4];
    const float* conv2_w = (const float*)d_in[5];
    const float* conv2_b = (const float*)d_in[6];
    const float* gn2_g   = (const float*)d_in[7];
    const float* gn2_b   = (const float*)d_in[8];
    const float* conv3_w = (const float*)d_in[9];
    const float* conv3_b = (const float*)d_in[10];
    const float* gn3_g   = (const float*)d_in[11];
    const float* gn3_b   = (const float*)d_in[12];
    const float* qw = (const float*)d_in[13];
    const float* qb = (const float*)d_in[14];
    const float* kw = (const float*)d_in[15];
    const float* kb = (const float*)d_in[16];
    const float* vw = (const float*)d_in[17];
    const float* vb = (const float*)d_in[18];
    const float* ln1_g = (const float*)d_in[19];
    const float* ln1_b = (const float*)d_in[20];
    const float* wi = (const float*)d_in[21];
    const float* bi = (const float*)d_in[22];
    const float* wo = (const float*)d_in[23];
    const float* bo = (const float*)d_in[24];
    const float* ln2_g = (const float*)d_in[25];
    const float* ln2_b = (const float*)d_in[26];
    const float* out_w = (const float*)d_in[27];
    const float* out_b = (const float*)d_in[28];

    const int B = 4, Cin = 12, L0 = 512000;
    const int C1 = 8,  L1 = 102400;
    const int C2 = 80, L2 = 10240;
    const int C3 = 800;
    const int S = 1024, D = 800, H = 8, Dh = 100, FF = 3200, LYR = 8;
    const int M = B * S;                       // 4096
    const long F = (long)B * S * D;            // 3,276,800 (== all conv stage sizes)

    // workspace carve-up
    char* ws = (char*)d_ws;
    size_t off = 0;
    auto carve = [&](size_t bytes) { char* p = ws + off; off += (bytes + 255) & ~(size_t)255; return p; };
    float* bufA = (float*)carve(F * 4);
    float* bufB = (float*)carve(F * 4);
    float* hbuf = (float*)carve(F * 4);
    float* qb_f = (float*)carve(F * 4);
    float* kb_f = (float*)carve(F * 4);
    float* vb_f = (float*)carve(F * 4);
    float* tmp  = (float*)carve(F * 4);
    unsigned short* hb16  = (unsigned short*)carve(F * 2);
    unsigned short* ffb16 = (unsigned short*)carve((size_t)M * FF * 2);
    unsigned short* wstg  = (unsigned short*)carve((size_t)FF * D * 2);
    float* stats = (float*)carve(64);

    const int eb = (int)((F + 255) / 256);     // elementwise blocks (12800)

    // ---- conv frontend ----
    conv1d_k<<<eb, 256, 0, stream>>>(src, conv1_w, conv1_b, bufA, B, Cin, L0, C1, L1, 201, 5, 100);
    gn_reduce_k<<<B, 256, 0, stream>>>(bufA, stats, (long)C1 * L1, 1e-5f);
    gn_apply_gelu_k<<<eb, 256, 0, stream>>>(bufA, stats, gn1_g, gn1_b, C1, L1, F);
    conv1d_k<<<eb, 256, 0, stream>>>(bufA, conv2_w, conv2_b, bufB, B, C1, L1, C2, L2, 101, 10, 50);
    gn_reduce_k<<<B, 256, 0, stream>>>(bufB, stats, (long)C2 * L2, 1e-5f);
    gn_apply_gelu_k<<<eb, 256, 0, stream>>>(bufB, stats, gn2_g, gn2_b, C2, L2, F);
    conv1d_k<<<eb, 256, 0, stream>>>(bufB, conv3_w, conv3_b, bufA, B, C2, L2, C3, S, 51, 10, 25);
    gn_reduce_k<<<B, 256, 0, stream>>>(bufA, stats, (long)C3 * S, 1e-5f);
    gn_apply_gelu_k<<<eb, 256, 0, stream>>>(bufA, stats, gn3_g, gn3_b, C3, S, F);
    transpose_pe_k<<<eb, 256, 0, stream>>>(bufA, hbuf, B, D, S);

    const dim3 gN800(M / 256, D / 32);         // 16 x 25
    const dim3 gN3200(M / 256, FF / 32);       // 16 x 100
    const int wb800  = (int)(((long)D * D + 255) / 256);
    const int wb3200 = (int)(((long)FF * D + 255) / 256);
    const float qscale = 0.1f;                 // 1/sqrt(Dh=100)

    // ---- transformer layers ----
    for (int l = 0; l < LYR; ++l) {
        const float* qw_l = qw + (size_t)l * D * D;
        const float* kw_l = kw + (size_t)l * D * D;
        const float* vw_l = vw + (size_t)l * D * D;
        const float* wi_l = wi + (size_t)l * FF * D;
        const float* wo_l = wo + (size_t)l * D * FF;

        tobf16_k<<<eb, 256, 0, stream>>>(hbuf, hb16, F);

        tobf16_k<<<wb800, 256, 0, stream>>>(qw_l, wstg, (long)D * D);
        gemm_bf16_wmma_k<<<gN800, 256, 0, stream>>>(hb16, wstg, qb + (size_t)l * D,
                                                    qb_f, nullptr, M, D, D, qscale, 0);
        tobf16_k<<<wb800, 256, 0, stream>>>(kw_l, wstg, (long)D * D);
        gemm_bf16_wmma_k<<<gN800, 256, 0, stream>>>(hb16, wstg, kb + (size_t)l * D,
                                                    kb_f, nullptr, M, D, D, 1.0f, 0);
        tobf16_k<<<wb800, 256, 0, stream>>>(vw_l, wstg, (long)D * D);
        gemm_bf16_wmma_k<<<gN800, 256, 0, stream>>>(hb16, wstg, vb + (size_t)l * D,
                                                    vb_f, nullptr, M, D, D, 1.0f, 0);

        attn_k<<<(B * H * (S / WATT) * WATT) / 8, 256, 0, stream>>>(qb_f, kb_f, vb_f, tmp, B, S, H, Dh);
        add_ln_k<<<M, 256, 0, stream>>>(hbuf, tmp, ln1_g + (size_t)l * D, ln1_b + (size_t)l * D, D);

        tobf16_k<<<eb, 256, 0, stream>>>(hbuf, hb16, F);
        tobf16_k<<<wb3200, 256, 0, stream>>>(wi_l, wstg, (long)FF * D);
        gemm_bf16_wmma_k<<<gN3200, 256, 0, stream>>>(hb16, wstg, bi + (size_t)l * FF,
                                                     nullptr, ffb16, M, FF, D, 1.0f, 1); // fused GELU -> bf16
        tobf16_k<<<wb3200, 256, 0, stream>>>(wo_l, wstg, (long)D * FF);
        gemm_bf16_wmma_k<<<gN800, 256, 0, stream>>>(ffb16, wstg, bo + (size_t)l * D,
                                                    tmp, nullptr, M, D, FF, 1.0f, 0);
        add_ln_k<<<M, 256, 0, stream>>>(hbuf, tmp, ln2_g + (size_t)l * D, ln2_b + (size_t)l * D, D);
    }

    // ---- output projection ----
    tobf16_k<<<eb, 256, 0, stream>>>(hbuf, hb16, F);
    tobf16_k<<<wb800, 256, 0, stream>>>(out_w, wstg, (long)D * D);
    gemm_bf16_wmma_k<<<gN800, 256, 0, stream>>>(hb16, wstg, out_b,
                                                (float*)d_out, nullptr, M, D, D, 1.0f, 0);
}
